// MemoryModule_17540646437554
// MI455X (gfx1250) — matrix operand order repaired and checked by reading, not compiled, not run on hardware
//
#include <hip/hip_runtime.h>

typedef _Float16 h16;
typedef __attribute__((ext_vector_type(4)))  _Float16 v4h;
typedef __attribute__((ext_vector_type(8)))  _Float16 v8h;
typedef __attribute__((ext_vector_type(16))) _Float16 v16h;
typedef __attribute__((ext_vector_type(8)))  float    v8f;

#define SX 776   // LDS row stride (halves) for X = [nf | h | ef], 768 + 8 pad
#define SM 264   // LDS row stride (halves) for m1/m, 256 + 8 pad

__device__ __forceinline__ v8f zerov() {
  v8f z = {0.f,0.f,0.f,0.f,0.f,0.f,0.f,0.f};
  return z;
}

__device__ __forceinline__ v16h cat16(v8h lo, v8h hi) {
  return __builtin_shufflevector(lo, hi, 0,1,2,3,4,5,6,7,8,9,10,11,12,13,14,15);
}

// A-fragment (16x32 f16) from LDS, row-major tile with given stride.
// ISA layout: lanes 0-15 row M=lane hold K = k0..k0+7 (VGPR0-3) and
// k0+16..k0+23 (VGPR4-7); lanes 16-31 hold K = k0+8.. and k0+24..
__device__ __forceinline__ v16h afrag(const h16* tile, int stride, int lane, int k0) {
  int r  = lane & 15;
  int kb = k0 + ((lane >> 4) << 3);
  const h16* p = tile + r * stride + kb;
  v8h lo = *(const v8h*)(p);
  v8h hi = *(const v8h*)(p + 16);
  return cat16(lo, hi);
}

// B-fragment (32x16 f16) from global packed weights stored [n][K] (column-major B).
// Lanes 0-15: col n, K = k0..k0+7, k0+16..k0+23; lanes 16-31: col n, K = k0+8..k0+15, k0+24..k0+31.
__device__ __forceinline__ v16h bfrag(const h16* w, int Kt, int n0, int lane, int k0) {
  int n  = n0 + (lane & 15);
  int kb = k0 + ((lane >> 4) << 3);
  const h16* p = w + (size_t)n * Kt + kb;
  v8h lo = *(const v8h*)(p);
  v8h hi = *(const v8h*)(p + 16);
  return cat16(lo, hi);
}

__device__ __forceinline__ v8f wmma16(v16h a, v16h b, v8f c) {
  return __builtin_amdgcn_wmma_f32_16x16x32_f16(false, a, false, b, (short)0, c, false, false);
}

__device__ __forceinline__ float sigmoidf_(float x) {
  return 1.0f / (1.0f + __expf(-x));
}

// Pack weights to f16 into d_ws:
//   W1p  [256][768]  = W1^T   (B for GEMM1)
//   W2p  [256][256]  = W2^T   (B for GEMM2)
//   Wihp [768][256]  = Wih    (rows are already B-columns for m @ Wih^T)
//   Whhp [768][256]  = Whh
__global__ void pack_weights(const float* __restrict__ W1, const float* __restrict__ W2,
                             const float* __restrict__ Wih, const float* __restrict__ Whh,
                             h16* __restrict__ ws) {
  h16* W1p  = ws;
  h16* W2p  = ws + 768 * 256;
  h16* Wihp = W2p + 256 * 256;
  h16* Whhp = Wihp + 768 * 256;
  int i = blockIdx.x * blockDim.x + threadIdx.x;
  if (i < 768 * 256) {
    int n = i / 768, k = i - n * 768;
    W1p[i]  = (h16)W1[k * 256 + n];
    Wihp[i] = (h16)Wih[i];
    Whhp[i] = (h16)Whh[i];
  }
  if (i < 256 * 256) {
    int n = i / 256, k = i - n * 256;
    W2p[i] = (h16)W2[k * 256 + n];
  }
}

__global__ __launch_bounds__(256)
void tgn_fused(const float* __restrict__ memory,
               const int*   __restrict__ node_idxs,
               const float* __restrict__ nodef,
               const float* __restrict__ edgef,
               const float* __restrict__ ts,
               const float* __restrict__ b1,
               const float* __restrict__ b2,
               const float* __restrict__ bih,
               const float* __restrict__ bhh,
               const h16*   __restrict__ ws,
               float* __restrict__ out_mem,
               float* __restrict__ out_msg,
               float* __restrict__ out_last,
               int Btot)
{
  extern __shared__ h16 smem[];
  h16* XA = smem;                 // [32][SX] f16: cols 0-255 nf, 256-511 h, 512-767 ef
  h16* MA = smem + 32 * SX;       // [32][SM] f16: m1 then m
  int* IDX = (int*)(smem + 32 * SX + 32 * SM);

  const h16* W1p  = ws;
  const h16* W2p  = ws + 768 * 256;
  const h16* Wihp = W2p + 256 * 256;
  const h16* Whhp = Wihp + 768 * 256;

  const int tid  = threadIdx.x;
  const int w    = tid >> 5;
  const int lane = tid & 31;
  const int row0 = blockIdx.x * 32;

  if (tid < 32) {
    int rg = row0 + tid;
    int rc = rg < Btot ? rg : (Btot - 1);
    int id = node_idxs[rc];
    IDX[tid] = id;
    if (rg < Btot) out_last[id] = ts[rg];
  }
  __syncthreads();

  // ---- Stage 1: build XA (f32 -> f16) ----
  for (int t = tid; t < 32 * 192; t += 256) {
    int r = t / 192;
    int c = (t - r * 192) << 2;
    int rg = row0 + r; if (rg >= Btot) rg = Btot - 1;
    const float* src;
    if (c < 256)      src = nodef  + (size_t)rg * 256 + c;
    else if (c < 512) src = memory + (size_t)IDX[r] * 256 + (c - 256);
    else              src = edgef  + (size_t)rg * 256 + (c - 512);
    float4 v = *(const float4*)src;
    v4h o = { (h16)v.x, (h16)v.y, (h16)v.z, (h16)v.w };
    *(v4h*)(XA + r * SX + c) = o;
  }
  __syncthreads();

  const int mrow = (lane >> 4) << 3;   // 0 or 8 (C/D layout row offset)
  const int nlan = lane & 15;

  // ---- GEMM1: m1 = relu(X @ W1 + b1), K = 768 ----
  #pragma unroll
  for (int ci = 0; ci < 2; ++ci) {
    int ct = w * 2 + ci;                  // 16-col tile, 0..15
    v8f acc0 = zerov(), acc1 = zerov();
    for (int k0 = 0; k0 < 768; k0 += 32) {
      v16h b  = bfrag(W1p, 768, ct * 16, lane, k0);
      v16h a0 = afrag(XA,            SX, lane, k0);
      v16h a1 = afrag(XA + 16 * SX,  SX, lane, k0);
      acc0 = wmma16(a0, b, acc0);
      acc1 = wmma16(a1, b, acc1);
    }
    int n = ct * 16 + nlan;
    float bb = b1[n];
    #pragma unroll
    for (int v2 = 0; v2 < 8; ++v2) {
      MA[(mrow + v2) * SM + n]      = (h16)fmaxf(acc0[v2] + bb, 0.f);
      MA[(16 + mrow + v2) * SM + n] = (h16)fmaxf(acc1[v2] + bb, 0.f);
    }
  }
  __syncthreads();

  // ---- GEMM2: m = m1 @ W2 + b2, K = 256 ----
  v8f macc[2][2];
  macc[0][0] = zerov(); macc[0][1] = zerov();
  macc[1][0] = zerov(); macc[1][1] = zerov();
  for (int k0 = 0; k0 < 256; k0 += 32) {
    v16h a0 = afrag(MA,            SM, lane, k0);
    v16h a1 = afrag(MA + 16 * SM,  SM, lane, k0);
    v16h bA = bfrag(W2p, 256, (w * 2) * 16,     lane, k0);
    v16h bB = bfrag(W2p, 256, (w * 2 + 1) * 16, lane, k0);
    macc[0][0] = wmma16(a0, bA, macc[0][0]);
    macc[0][1] = wmma16(a1, bA, macc[0][1]);
    macc[1][0] = wmma16(a0, bB, macc[1][0]);
    macc[1][1] = wmma16(a1, bB, macc[1][1]);
  }
  __syncthreads();   // all m1 reads done before MA is overwritten with m
  #pragma unroll
  for (int ci = 0; ci < 2; ++ci) {
    int n = (w * 2 + ci) * 16 + nlan;
    float bb = b2[n];
    #pragma unroll
    for (int rt = 0; rt < 2; ++rt) {
      #pragma unroll
      for (int v2 = 0; v2 < 8; ++v2) {
        int row = rt * 16 + mrow + v2;
        float val = macc[ci][rt][v2] + bb;
        MA[row * SM + n] = (h16)val;
        if (row0 + row < Btot)
          out_msg[(size_t)IDX[row] * 256 + n] = val;
      }
    }
  }
  __syncthreads();

  // ---- GRU gates + update ----
  // r/z: (gx+gh) = [m,h] @ [Wih,Whh]^T  (K=512); n-gate: xn, hn kept separate.
  const h16* HA = XA + 256;   // h as f16, stride SX
  #pragma unroll 1
  for (int ci = 0; ci < 2; ++ci) {
    int cg = w * 2 + ci;
    int nr = cg * 16;         // gate-column base within [0,256)
    v8f gr[2], gz[2], gxn[2], ghn[2];
    gr[0] = zerov(); gr[1] = zerov(); gz[0] = zerov(); gz[1] = zerov();
    gxn[0] = zerov(); gxn[1] = zerov(); ghn[0] = zerov(); ghn[1] = zerov();

    for (int k0 = 0; k0 < 512; k0 += 32) {
      v16h a0, a1, br, bz;
      if (k0 < 256) {
        a0 = afrag(MA,           SM, lane, k0);
        a1 = afrag(MA + 16 * SM, SM, lane, k0);
        br = bfrag(Wihp, 256, nr,       lane, k0);
        bz = bfrag(Wihp, 256, 256 + nr, lane, k0);
      } else {
        a0 = afrag(HA,           SX, lane, k0 - 256);
        a1 = afrag(HA + 16 * SX, SX, lane, k0 - 256);
        br = bfrag(Whhp, 256, nr,       lane, k0 - 256);
        bz = bfrag(Whhp, 256, 256 + nr, lane, k0 - 256);
      }
      gr[0] = wmma16(a0, br, gr[0]); gr[1] = wmma16(a1, br, gr[1]);
      gz[0] = wmma16(a0, bz, gz[0]); gz[1] = wmma16(a1, bz, gz[1]);
    }
    for (int k0 = 0; k0 < 256; k0 += 32) {
      v16h a0 = afrag(MA,           SM, lane, k0);
      v16h a1 = afrag(MA + 16 * SM, SM, lane, k0);
      v16h b  = bfrag(Wihp, 256, 512 + nr, lane, k0);
      gxn[0] = wmma16(a0, b, gxn[0]); gxn[1] = wmma16(a1, b, gxn[1]);
    }
    for (int k0 = 0; k0 < 256; k0 += 32) {
      v16h a0 = afrag(HA,           SX, lane, k0);
      v16h a1 = afrag(HA + 16 * SX, SX, lane, k0);
      v16h b  = bfrag(Whhp, 256, 512 + nr, lane, k0);
      ghn[0] = wmma16(a0, b, ghn[0]); ghn[1] = wmma16(a1, b, ghn[1]);
    }

    int c = nr + nlan;
    float br_ = bih[c]       + bhh[c];
    float bz_ = bih[256 + c] + bhh[256 + c];
    float bxn = bih[512 + c];
    float bhn = bhh[512 + c];
    #pragma unroll
    for (int rt = 0; rt < 2; ++rt) {
      #pragma unroll
      for (int v2 = 0; v2 < 8; ++v2) {
        int row = rt * 16 + mrow + v2;
        int rg  = row0 + row;
        float r  = sigmoidf_(gr[rt][v2] + br_);
        float z  = sigmoidf_(gz[rt][v2] + bz_);
        float nn = tanhf(gxn[rt][v2] + bxn + r * (ghn[rt][v2] + bhn));
        size_t off = (size_t)IDX[row] * 256 + c;
        float hold = memory[off];
        float hnew = (1.f - z) * nn + z * hold;
        if (rg < Btot) out_mem[off] = hnew;
      }
    }
  }
}

extern "C" void kernel_launch(void* const* d_in, const int* in_sizes, int n_in,
                              void* d_out, int out_size, void* d_ws, size_t ws_size,
                              hipStream_t stream) {
  const float* memory = (const float*)d_in[0];
  const float* msgs   = (const float*)d_in[1];
  const float* lastu  = (const float*)d_in[2];
  const int*   idxs   = (const int*)d_in[3];
  const float* nodef  = (const float*)d_in[4];
  const float* edgef  = (const float*)d_in[5];
  const float* ts     = (const float*)d_in[6];
  const float* W1  = (const float*)d_in[7];
  const float* b1  = (const float*)d_in[8];
  const float* W2  = (const float*)d_in[9];
  const float* b2  = (const float*)d_in[10];
  const float* Wih = (const float*)d_in[11];
  const float* Whh = (const float*)d_in[12];
  const float* bih = (const float*)d_in[13];
  const float* bhh = (const float*)d_in[14];

  size_t memN  = (size_t)in_sizes[0];   // N_NODES*MEM
  size_t msgN  = (size_t)in_sizes[1];   // N_NODES*MSG
  size_t lastN = (size_t)in_sizes[2];   // N_NODES
  int Btot = in_sizes[3];

  float* out_mem  = (float*)d_out;
  float* out_msg  = out_mem + memN;
  float* out_last = out_msg + msgN;

  // Base-table pass-through (HBM streaming copies).
  hipMemcpyAsync(out_mem,  memory, memN  * sizeof(float), hipMemcpyDeviceToDevice, stream);
  hipMemcpyAsync(out_msg,  msgs,   msgN  * sizeof(float), hipMemcpyDeviceToDevice, stream);
  hipMemcpyAsync(out_last, lastu,  lastN * sizeof(float), hipMemcpyDeviceToDevice, stream);

  // Pack weights to f16 (transposed for W1/W2) into workspace.
  h16* ws = (h16*)d_ws;
  {
    int total = 768 * 256;
    pack_weights<<<(total + 255) / 256, 256, 0, stream>>>(W1, W2, Wih, Whh, ws);
  }

  // Fused MLP + GRU + scatter.
  {
    int blocks = (Btot + 31) / 32;
    size_t shmem = (size_t)(32 * SX + 32 * SM) * sizeof(h16) + 32 * sizeof(int);
    tgn_fused<<<blocks, 256, shmem, stream>>>(memory, idxs, nodef, edgef, ts,
                                              b1, b2, bih, bhh, ws,
                                              out_mem, out_msg, out_last, Btot);
  }
}
